// MultiStreamMambaEncoder_27986006900762
// MI455X (gfx1250) — compile-verified, matrix operand order checked by reading, not run on hardware
//
#include <hip/hip_runtime.h>

typedef __attribute__((ext_vector_type(16))) _Float16 v16h;
typedef __attribute__((ext_vector_type(8)))  _Float16 v8h;
typedef __attribute__((ext_vector_type(8)))  float    v8f;
typedef __attribute__((ext_vector_type(4)))  int      v4i;

#define DMODEL 128
#define DINNER 256
#define DSTATE 16
#define DTRANK 8
#define DBCDIM 40   // DTRANK + 2*DSTATE

// CDNA5 async global->LDS path (guarded: falls back to plain copies if the
// toolchain lacks the builtin, so compilation never breaks).
// Probe-confirmed signature: (v4i AS1*, v4i AS3*, imm offset, imm cpol).
#if defined(__AMDGCN__) && __has_builtin(__builtin_amdgcn_global_load_async_to_lds_b128)
#define HAVE_ASYNC_LDS 1
#else
#define HAVE_ASYNC_LDS 0
#endif

// ---------------------------------------------------------------------------
// embed: out[row, c] = sum_f in[row, f] * W[f, c] + b[c]
// ---------------------------------------------------------------------------
__global__ void embed_kernel(const float* __restrict__ in, const float* __restrict__ W,
                             const float* __restrict__ bias, float* __restrict__ out, int F) {
    int row = blockIdx.x;
    int c   = threadIdx.x;
    const float* ip = in + (size_t)row * F;
    float acc = bias[c];
    for (int f = 0; f < F; ++f) acc += ip[f] * W[f * DMODEL + c];
    out[(size_t)row * DMODEL + c] = acc;
}

// ---------------------------------------------------------------------------
// rmsnorm over 128 cols; writes f32 (for final mean) and f16 (GEMM A operand)
// ---------------------------------------------------------------------------
__global__ void rmsnorm_kernel(const float* __restrict__ x, const float* __restrict__ w,
                               float* __restrict__ outf, _Float16* __restrict__ outh) {
    __shared__ float red[DMODEL];
    int row = blockIdx.x;
    int c   = threadIdx.x;
    float v = x[(size_t)row * DMODEL + c];
    red[c] = v * v;
    __syncthreads();
    for (int s = 64; s > 0; s >>= 1) {
        if (c < s) red[c] += red[c + s];
        __syncthreads();
    }
    float scale = rsqrtf(red[0] * (1.0f / DMODEL) + 1e-5f);
    float o = v * scale * w[c];
    outf[(size_t)row * DMODEL + c] = o;
    outh[(size_t)row * DMODEL + c] = (_Float16)o;
}

// ---------------------------------------------------------------------------
// weight prep: Wt[n*K + k] = (f16) W[k*N + n]   (transpose-convert, once/layer)
// ---------------------------------------------------------------------------
__global__ void wprep_kernel(const float* __restrict__ W, _Float16* __restrict__ Wt,
                             int K, int N) {
    int i = blockIdx.x * 256 + threadIdx.x;
    if (i >= K * N) return;
    int n = i / K, k = i - n * K;
    Wt[i] = (_Float16)W[(size_t)k * N + n];
}

// ---------------------------------------------------------------------------
// WMMA GEMM: C[M,N] = A[M,K]f16 @ Wt[N,K]f16^T (+ Res), f32 accumulate.
// Workgroup = 8 waves -> 128(M) x 64(N) slab. The K x 64 weight slab is
// async-copied global->LDS (b128) once and reused by all waves. Each wave
// holds 4 accumulators (16x64), reusing its A fragment 4x per k-step.
// Fragment layouts per CDNA5 ISA: lane l -> M(or N) = l&15, K halves
// [kb, kb+8) and [kb+16, kb+24) with kb = k0 + 8*(l>>4); C/D: lane l holds
// N = l&15, M = v + 8*(l>>4) in VGPR v.
// ---------------------------------------------------------------------------
template <int K, bool RES>
__global__ __launch_bounds__(256) void gemm_wmma(
    const _Float16* __restrict__ A,   // [M][K]
    const _Float16* __restrict__ Bt,  // [N][K]
    float* __restrict__ C,            // [M][N]
    const float* __restrict__ Res,    // [M][N] (used iff RES)
    int M, int N) {
    __shared__ _Float16 Blds[64 * K];

    int tid  = threadIdx.x;
    int wave = tid >> 5;
    int lane = tid & 31;
    int l15  = lane & 15;
    int hs   = lane >> 4;
    int tm   = blockIdx.x;
    int n0   = blockIdx.y * 64;

    // ---- stage 64 x K f16 weight slab into LDS (rows contiguous) ----
    const int CHUNKS = 64 * K / 8;            // 16-byte transfers
#pragma unroll
    for (int i = tid; i < CHUNKS; i += 256) {
        int r = i / (K / 8);
        int c = i - r * (K / 8);
        size_t goff = (size_t)(n0 + r) * K + c * 8;
        int    loff = r * K + c * 8;
#if HAVE_ASYNC_LDS
        __builtin_amdgcn_global_load_async_to_lds_b128(
            (__attribute__((address_space(1))) v4i*)(Bt + goff),
            (__attribute__((address_space(3))) v4i*)&Blds[loff], 0, 0);
#else
        *(ulong2*)&Blds[loff] = *(const ulong2*)(Bt + goff);
#endif
    }
#if HAVE_ASYNC_LDS
#if __has_builtin(__builtin_amdgcn_s_wait_asynccnt)
    __builtin_amdgcn_s_wait_asynccnt(0);
#else
    asm volatile("s_wait_asynccnt 0" ::: "memory");
#endif
#endif
    __syncthreads();

    // ---- main loop ----
    int mrow = tm * 128 + wave * 16 + l15;
    if (mrow >= M) mrow = M - 1;              // clamp: EXEC stays all-1s
    v8f acc[4] = {};
#pragma unroll
    for (int k0 = 0; k0 < K; k0 += 32) {
        int kb = k0 + hs * 8;
        const _Float16* ap = A + (size_t)mrow * K + kb;
        v8h alo = *(const v8h*)(ap);
        v8h ahi = *(const v8h*)(ap + 16);
        v16h a = __builtin_shufflevector(alo, ahi, 0, 1, 2, 3, 4, 5, 6, 7,
                                         8, 9, 10, 11, 12, 13, 14, 15);
#pragma unroll
        for (int j = 0; j < 4; ++j) {
            const _Float16* bp = &Blds[(j * 16 + l15) * K + kb];
            v8h blo = *(const v8h*)(bp);
            v8h bhi = *(const v8h*)(bp + 16);
            v16h b = __builtin_shufflevector(blo, bhi, 0, 1, 2, 3, 4, 5, 6, 7,
                                             8, 9, 10, 11, 12, 13, 14, 15);
            acc[j] = __builtin_amdgcn_wmma_f32_16x16x32_f16(
                false, a, false, b, (short)0, acc[j], false, false);
        }
    }

    // ---- epilogue (straight-line; RES is a compile-time flag) ----
#pragma unroll
    for (int j = 0; j < 4; ++j) {
        int nc = n0 + j * 16 + l15;
#pragma unroll
        for (int v = 0; v < 8; ++v) {
            int mr = tm * 128 + wave * 16 + hs * 8 + v;
            if (mr < M) {
                float val = acc[j][v];
                if (RES) val += Res[(size_t)mr * N + nc];
                C[(size_t)mr * N + nc] = val;
            }
        }
    }
}

// ---------------------------------------------------------------------------
// causal depthwise conv (K=4) on h-half of hg, + bias, + SiLU
// ---------------------------------------------------------------------------
__global__ void conv_silu_kernel(const float* __restrict__ hg, const float* __restrict__ cw,
                                 const float* __restrict__ cb, float* __restrict__ h, int L) {
    int row = blockIdx.x;            // b*L + t
    int c   = threadIdx.x;           // 0..255
    int t   = row % L;
    float acc = cb[c];
#pragma unroll
    for (int k = 0; k < 4; ++k) {
        int tt = t - 3 + k;
        if (tt >= 0) acc += hg[(size_t)(row - 3 + k) * 512 + c] * cw[c * 4 + k];
    }
    h[(size_t)row * DINNER + c] = acc / (1.0f + __expf(-acc));
}

// ---------------------------------------------------------------------------
// x_proj: dbc[row, j] = sum_k h[row, k] * xw[k, j]
// ---------------------------------------------------------------------------
__global__ void xproj_kernel(const float* __restrict__ h, const float* __restrict__ xw,
                             float* __restrict__ dbc) {
    int row = blockIdx.x;
    int j   = threadIdx.x;
    if (j >= DBCDIM) return;
    const float* hp = h + (size_t)row * DINNER;
    float acc = 0.f;
    for (int k = 0; k < DINNER; ++k) acc += hp[k] * xw[k * DBCDIM + j];
    dbc[(size_t)row * DBCDIM + j] = acc;
}

// ---------------------------------------------------------------------------
// selective scan: fused softplus(dt) + SSM recurrence + gating, y in f16.
// One block per batch, one thread per channel, 16 states in VGPRs.
// ---------------------------------------------------------------------------
__global__ void scan_kernel(const float* __restrict__ h, const float* __restrict__ hg,
                            const float* __restrict__ dbc,
                            const float* __restrict__ dt_w, const float* __restrict__ dt_b,
                            const float* __restrict__ A_log, const float* __restrict__ Dp,
                            _Float16* __restrict__ y, int L) {
    int b = blockIdx.x;
    int d = threadIdx.x;             // 0..255
    __shared__ float sBC[DBCDIM];
    float A[DSTATE], st[DSTATE];
#pragma unroll
    for (int n = 0; n < DSTATE; ++n) {
        A[n]  = -__expf(A_log[d * DSTATE + n]);
        st[n] = 0.f;
    }
    float dtw[DTRANK];
#pragma unroll
    for (int r = 0; r < DTRANK; ++r) dtw[r] = dt_w[r * DINNER + d];
    float dtb = dt_b[d];
    float Dd  = Dp[d];

    for (int t = 0; t < L; ++t) {
        size_t idx = (size_t)b * L + t;
        if (d < DBCDIM) sBC[d] = dbc[idx * DBCDIM + d];
        __syncthreads();
        float dt = dtb;
#pragma unroll
        for (int r = 0; r < DTRANK; ++r) dt += sBC[r] * dtw[r];
        dt = (dt > 20.f) ? dt : __logf(1.f + __expf(dt));   // softplus
        float u  = h[idx * DINNER + d];
        float du = dt * u;
        float yv = 0.f;
#pragma unroll
        for (int n = 0; n < DSTATE; ++n) {
            st[n] = st[n] * __expf(dt * A[n]) + du * sBC[DTRANK + n];
            yv   += st[n] * sBC[DTRANK + DSTATE + n];
        }
        yv += u * Dd;
        float g = hg[idx * 512 + DINNER + d];
        y[idx * DINNER + d] = (_Float16)(yv * (g / (1.f + __expf(-g))));
        __syncthreads();
    }
}

// ---------------------------------------------------------------------------
// mean over sequence: out[b*stride + c] = mean_t xn[b, t, c]
// ---------------------------------------------------------------------------
__global__ void mean_kernel(const float* __restrict__ xn, float* __restrict__ out,
                            int L, int out_stride) {
    int b = blockIdx.x;
    int c = threadIdx.x;
    float acc = 0.f;
    for (int t = 0; t < L; ++t) acc += xn[((size_t)b * L + t) * DMODEL + c];
    out[(size_t)b * out_stride + c] = acc / (float)L;
}

// ---------------------------------------------------------------------------
// Host-side: run one 2-layer Mamba encoder in place on x[Bn*L, 128].
// pp -> 21 tensors: layer0(10), layer1(10), norm_f; layer order:
// norm_w, in_proj, conv_w, conv_b, x_proj, dt_w, dt_b, A_log, D, out_proj
// ---------------------------------------------------------------------------
static void run_encoder(const float* const* pp, float* x, int Bn, int L,
                        float* xnf, _Float16* xn16, float* hg, float* h, float* dbc,
                        _Float16* y16, _Float16* wt16,
                        float* feat_out, int feat_stride, hipStream_t stream) {
    int rows  = Bn * L;
    int gridM = (rows + 127) / 128;
    for (int l = 0; l < 2; ++l) {
        const float* const* q = pp + l * 10;
        rmsnorm_kernel<<<rows, DMODEL, 0, stream>>>(x, q[0], xnf, xn16);
        wprep_kernel<<<(DMODEL * 512 + 255) / 256, 256, 0, stream>>>(q[1], wt16, DMODEL, 512);
        gemm_wmma<DMODEL, false><<<dim3(gridM, 512 / 64), 256, 0, stream>>>(
            xn16, wt16, hg, nullptr, rows, 512);
        conv_silu_kernel<<<rows, DINNER, 0, stream>>>(hg, q[2], q[3], h, L);
        xproj_kernel<<<rows, 64, 0, stream>>>(h, q[4], dbc);
        scan_kernel<<<Bn, DINNER, 0, stream>>>(h, hg, dbc, q[5], q[6], q[7], q[8], y16, L);
        wprep_kernel<<<(DINNER * DMODEL + 255) / 256, 256, 0, stream>>>(q[9], wt16, DINNER, DMODEL);
        gemm_wmma<DINNER, true><<<dim3(gridM, DMODEL / 64), 256, 0, stream>>>(
            y16, wt16, x, x, rows, DMODEL);
    }
    rmsnorm_kernel<<<rows, DMODEL, 0, stream>>>(x, pp[20], xnf, xn16);
    mean_kernel<<<Bn, DMODEL, 0, stream>>>(xnf, feat_out, L, feat_stride);
}

// ---------------------------------------------------------------------------
// kernel_launch
// ---------------------------------------------------------------------------
extern "C" void kernel_launch(void* const* d_in, const int* in_sizes, int n_in,
                              void* d_out, int out_size, void* d_ws, size_t ws_size,
                              hipStream_t stream) {
    (void)in_sizes; (void)n_in; (void)out_size; (void)ws_size;

    const float* ego     = (const float*)d_in[0];   // (16,512,6)
    const float* soc     = (const float*)d_in[1];   // (16,512,6)
    const float* mapt    = (const float*)d_in[2];   // (16,1024,3)
    const float* agent_w = (const float*)d_in[3];
    const float* agent_b = (const float*)d_in[4];
    const float* map_w   = (const float*)d_in[5];
    const float* map_b   = (const float*)d_in[6];
    const float* const* Pt = (const float* const*)&d_in[7];        // p_temporal
    const float* const* Pr = (const float* const*)&d_in[7 + 21];   // p_traffic
    const float* const* Ps = (const float* const*)&d_in[7 + 42];   // p_scene
    const float* const* Pg = (const float* const*)&d_in[7 + 63];   // p_global

    // workspace layout, sized for the largest encoder (scene: 16384 rows)
    const size_t R = 16384;
    float* ws   = (float*)d_ws;
    float* hg   = ws;                     // R*512 f32  (h | gate)
    float* h    = hg  + R * 512;          // R*256 f32
    float* dbc  = h   + R * 256;          // R*40  f32
    float* x    = dbc + R * 40;           // R*128 f32  (per-stream activations)
    float* xnf  = x   + R * 128;          // R*128 f32  (normed, for mean)
    float* gx   = xnf + R * 128;          // 48*128 f32 (stacked feats -> global enc)
    _Float16* xn16 = (_Float16*)(gx + 48 * 128);  // R*128 f16
    _Float16* y16  = xn16 + R * 128;              // R*256 f16
    _Float16* wt16 = y16  + R * 256;              // 65536 f16 (transposed weights)

    // --- three streams; features land directly in gx slots ---
    embed_kernel<<<16 * 512, DMODEL, 0, stream>>>(ego, agent_w, agent_b, x, 6);
    run_encoder(Pt, x, 16, 512, xnf, xn16, hg, h, dbc, y16, wt16,
                gx + 0 * DMODEL, 3 * DMODEL, stream);

    embed_kernel<<<16 * 512, DMODEL, 0, stream>>>(soc, agent_w, agent_b, x, 6);
    run_encoder(Pr, x, 16, 512, xnf, xn16, hg, h, dbc, y16, wt16,
                gx + 1 * DMODEL, 3 * DMODEL, stream);

    embed_kernel<<<16 * 1024, DMODEL, 0, stream>>>(mapt, map_w, map_b, x, 3);
    run_encoder(Ps, x, 16, 1024, xnf, xn16, hg, h, dbc, y16, wt16,
                gx + 2 * DMODEL, 3 * DMODEL, stream);

    // --- global encoder over stacked (16, 3, 128), mean -> d_out (16,128) ---
    run_encoder(Pg, gx, 16, 3, xnf, xn16, hg, h, dbc, y16, wt16,
                (float*)d_out, DMODEL, stream);
}
